// LinearBase_67860483277526
// MI455X (gfx1250) — matrix operand order repaired and checked by reading, not compiled, not run on hardware
//
#include <hip/hip_runtime.h>

// Problem dims (match reference)
#define M_DIM 8192
#define K_DIM 4096
#define H_DIM 16384

// GEMM tiling
#define BM 128
#define BN 128
#define BK 32
#define LDSTR 40   // LDS row stride in halfwords (80B = 20 banks, 16B aligned)

typedef __attribute__((ext_vector_type(16))) __bf16 v16bf;
typedef __attribute__((ext_vector_type(8)))  float  v8f;

typedef __attribute__((address_space(3))) unsigned short* lds_us_ptr;

union FragAB {
  uint4 u[2];
  v16bf bf;
};

// ---- bf16 hi/lo split helpers (truncation split; |err| <= 2^-16 relative) ----

__device__ __forceinline__ unsigned int pack_hi16(unsigned int u0, unsigned int u1) {
  return __builtin_amdgcn_perm(u1, u0, 0x07060302u);  // {u1.hi16, u0.hi16}
}

__device__ __forceinline__ void split2(float f0, float f1,
                                       unsigned int& hp, unsigned int& lp) {
  const unsigned int u0 = __float_as_uint(f0);
  const unsigned int u1 = __float_as_uint(f1);
  hp = pack_hi16(u0, u1);
  const float l0 = f0 - __uint_as_float(u0 & 0xFFFF0000u);
  const float l1 = f1 - __uint_as_float(u1 & 0xFFFF0000u);
  lp = pack_hi16(__float_as_uint(l0), __float_as_uint(l1));
}

__device__ __forceinline__ void split_bf16_t(float f, unsigned short& h, unsigned short& l) {
  const unsigned int u = __float_as_uint(f);
  h = (unsigned short)(u >> 16);
  const float r = f - __uint_as_float(u & 0xFFFF0000u);
  l = (unsigned short)(__float_as_uint(r) >> 16);
}

__device__ __forceinline__ v8f wmma_bf16(v16bf a, v16bf b, v8f c) {
  // 8 args: (neg_a, A, neg_b, B, c_mod, C, reuse_a, reuse_b)
  return __builtin_amdgcn_wmma_f32_16x16x32_bf16(false, a, false, b, (short)0, c, false, false);
}

// ---- gfx1250 async global->LDS copy (per-lane 16B, tracked by ASYNCcnt) ----
__device__ __forceinline__ void async_copy_b128(unsigned short* lds_p,
                                                const unsigned short* g_p) {
  lds_us_ptr l = (lds_us_ptr)lds_p;
  asm volatile("global_load_async_to_lds_b128 %0, %1, off"
               :: "v"(l), "v"(g_p) : "memory");
}

__device__ __forceinline__ void wait_async0() {
  asm volatile("s_wait_asynccnt 0x0" ::: "memory");
}

// ============================================================================
// Pass 1a: x[M,K] f32 -> x_hi[M,K], x_lo[M,K] bf16 (same layout)
// ============================================================================
__global__ __launch_bounds__(256)
void split_copy_kernel(const float* __restrict__ src,
                       unsigned short* __restrict__ hi,
                       unsigned short* __restrict__ lo) {
  const size_t i = (size_t)blockIdx.x * 256 + threadIdx.x;   // float4 index
  const float4 v = reinterpret_cast<const float4*>(src)[i];
  unsigned int hp0, lp0, hp1, lp1;
  split2(v.x, v.y, hp0, lp0);
  split2(v.z, v.w, hp1, lp1);
  reinterpret_cast<uint2*>(hi)[i] = make_uint2(hp0, hp1);
  reinterpret_cast<uint2*>(lo)[i] = make_uint2(lp0, lp1);
}

// ============================================================================
// Pass 1b: w[K,H] f32 -> wt_hi[H,K], wt_lo[H,K] bf16 (transposed, 32x32 tiles)
// ============================================================================
__global__ __launch_bounds__(256)
void split_transpose_kernel(const float* __restrict__ w,
                            unsigned short* __restrict__ hi,
                            unsigned short* __restrict__ lo) {
  __shared__ __align__(16) float tile[32 * 36];
  const int h0 = blockIdx.x * 32;
  const int k0 = blockIdx.y * 32;
  const int tr = threadIdx.x >> 3;         // 0..31
  const int tc = (threadIdx.x & 7) << 2;   // 0..28 step 4

  const float4 v = *reinterpret_cast<const float4*>(
      w + (size_t)(k0 + tr) * H_DIM + h0 + tc);
  *reinterpret_cast<float4*>(&tile[tr * 36 + tc]) = v;
  __syncthreads();

  const float f0 = tile[(tc + 0) * 36 + tr];
  const float f1 = tile[(tc + 1) * 36 + tr];
  const float f2 = tile[(tc + 2) * 36 + tr];
  const float f3 = tile[(tc + 3) * 36 + tr];
  unsigned int hp0, lp0, hp1, lp1;
  split2(f0, f1, hp0, lp0);
  split2(f2, f3, hp1, lp1);
  const size_t o = ((size_t)(h0 + tr) * K_DIM + k0 + tc) >> 2;  // uint2 index
  reinterpret_cast<uint2*>(hi)[o] = make_uint2(hp0, hp1);
  reinterpret_cast<uint2*>(lo)[o] = make_uint2(lp0, lp1);
}

// ============================================================================
// Pass 2 (fast path): GEMM on pre-split bf16.
// Staging via GLOBAL_LOAD_ASYNC_TO_LDS_B128 (ASYNCcnt), double-buffered LDS.
// A = x_hi/lo [M,K]; B = wt_hi/lo [H,K] (both K-contiguous).
// ============================================================================
__global__ __launch_bounds__(256, 1)
void gemm_pre_split_async_kernel(const unsigned short* __restrict__ xhi,
                                 const unsigned short* __restrict__ xlo,
                                 const unsigned short* __restrict__ whi,
                                 const unsigned short* __restrict__ wlo,
                                 const float* __restrict__ bias,
                                 float* __restrict__ out) {
  __shared__ __align__(16) unsigned short lds_a_hi[2][BM * LDSTR];
  __shared__ __align__(16) unsigned short lds_a_lo[2][BM * LDSTR];
  __shared__ __align__(16) unsigned short lds_b_hi[2][BN * LDSTR];  // [n][k]
  __shared__ __align__(16) unsigned short lds_b_lo[2][BN * LDSTR];

  const int tid    = threadIdx.x;
  const int lane   = tid & 31;
  const int wid    = tid >> 5;
  const int wave_m = wid >> 2;     // 0..1
  const int wave_n = wid & 3;      // 0..3
  const int msub   = lane & 15;
  const int ksel   = lane >> 4;

  const int bm = blockIdx.x * BM;
  const int bn = blockIdx.y * BN;

  // Staging: tile is 128 rows x 32 bf16 = 512 x 16B chunks; 2 chunks per thread.
  const int r0 = tid >> 2;                 // 0..63
  const int c0 = (tid & 3) << 3;           // 0,8,16,24 (bf16 units)
  const int r1 = r0 + 64;
  const size_t aBase = (size_t)bm * K_DIM;
  const size_t bBase = (size_t)bn * K_DIM;

  // Per-thread LDS destinations (buffer 0); buffer 1 = +BM*LDSTR halfwords ([BUF] index).
  const int ro0 = r0 * LDSTR + c0;
  const int ro1 = r1 * LDSTR + c0;

  v8f acc[4][2];
#pragma unroll
  for (int i = 0; i < 4; ++i)
#pragma unroll
    for (int j = 0; j < 2; ++j)
#pragma unroll
      for (int e = 0; e < 8; ++e) acc[i][j][e] = 0.0f;

#define ASYNC_TILE(KT, BUF)                                                       \
  do {                                                                            \
    const size_t ka0 = aBase + (size_t)r0 * K_DIM + (size_t)(KT) + c0;            \
    const size_t ka1 = aBase + (size_t)r1 * K_DIM + (size_t)(KT) + c0;            \
    const size_t kb0 = bBase + (size_t)r0 * K_DIM + (size_t)(KT) + c0;            \
    const size_t kb1 = bBase + (size_t)r1 * K_DIM + (size_t)(KT) + c0;            \
    async_copy_b128(&lds_a_hi[BUF][ro0], xhi + ka0);                              \
    async_copy_b128(&lds_a_hi[BUF][ro1], xhi + ka1);                              \
    async_copy_b128(&lds_a_lo[BUF][ro0], xlo + ka0);                              \
    async_copy_b128(&lds_a_lo[BUF][ro1], xlo + ka1);                              \
    async_copy_b128(&lds_b_hi[BUF][ro0], whi + kb0);                              \
    async_copy_b128(&lds_b_hi[BUF][ro1], whi + kb1);                              \
    async_copy_b128(&lds_b_lo[BUF][ro0], wlo + kb0);                              \
    async_copy_b128(&lds_b_lo[BUF][ro1], wlo + kb1);                              \
  } while (0)

  ASYNC_TILE(0, 0);

  const int NK = K_DIM / BK;   // 128
  int cur = 0;
  for (int it = 0; it < NK; ++it) {
    wait_async0();     // this wave's copies for tile `it` have landed in LDS
    __syncthreads();   // all waves' copies landed; all waves done reading buf cur^1

    if (it + 1 < NK) ASYNC_TILE((it + 1) * BK, cur ^ 1);  // overlap with WMMAs

    // ---- B fragments: 16 contiguous K-values per lane ----
    const unsigned short* bh = lds_b_hi[cur];
    const unsigned short* bl = lds_b_lo[cur];
    FragAB bhi[2], blo[2];
#pragma unroll
    for (int fn = 0; fn < 2; ++fn) {
      const int o = ((wave_n * 2 + fn) * 16 + msub) * LDSTR + ksel * 16;
      bhi[fn].u[0] = *reinterpret_cast<const uint4*>(&bh[o]);
      bhi[fn].u[1] = *reinterpret_cast<const uint4*>(&bh[o + 8]);
      blo[fn].u[0] = *reinterpret_cast<const uint4*>(&bl[o]);
      blo[fn].u[1] = *reinterpret_cast<const uint4*>(&bl[o + 8]);
    }

    // ---- A fragments: two 8-elem K chunks per lane; 3 WMMAs each ----
    const unsigned short* ah = lds_a_hi[cur];
    const unsigned short* al = lds_a_lo[cur];
#pragma unroll
    for (int fm = 0; fm < 4; ++fm) {
      const int o0 = ((wave_m * 4 + fm) * 16 + msub) * LDSTR + ksel * 16;
      FragAB ahi, alo;
      ahi.u[0] = *reinterpret_cast<const uint4*>(&ah[o0]);
      ahi.u[1] = *reinterpret_cast<const uint4*>(&ah[o0 + 8]);
      alo.u[0] = *reinterpret_cast<const uint4*>(&al[o0]);
      alo.u[1] = *reinterpret_cast<const uint4*>(&al[o0 + 8]);
#pragma unroll
      for (int fn = 0; fn < 2; ++fn) {
        acc[fm][fn] = wmma_bf16(ahi.bf, bhi[fn].bf, acc[fm][fn]);
        acc[fm][fn] = wmma_bf16(alo.bf, bhi[fn].bf, acc[fm][fn]);
        acc[fm][fn] = wmma_bf16(ahi.bf, blo[fn].bf, acc[fm][fn]);
      }
    }
    cur ^= 1;
  }
#undef ASYNC_TILE

  // ---- Epilogue: bias add + store ----
#pragma unroll
  for (int fm = 0; fm < 4; ++fm) {
#pragma unroll
    for (int fn = 0; fn < 2; ++fn) {
      const int col = bn + (wave_n * 2 + fn) * 16 + msub;
      const float bv = bias[col];
#pragma unroll
      for (int j = 0; j < 8; ++j) {
        const int row = bm + (wave_m * 4 + fm) * 16 + ksel * 8 + j;
        out[(size_t)row * H_DIM + col] = acc[fm][fn][j] + bv;
      }
    }
  }
}

// ============================================================================
// Fallback (ws too small): single-pass, convert during staging (perm split)
// ============================================================================
__global__ __launch_bounds__(256, 1)
void linear_bias_bf16x3_kernel(const float* __restrict__ x,
                               const float* __restrict__ w,
                               const float* __restrict__ bias,
                               float* __restrict__ out) {
  __shared__ __align__(16) unsigned short lds_a_hi[BM * LDSTR];
  __shared__ __align__(16) unsigned short lds_a_lo[BM * LDSTR];
  __shared__ __align__(16) unsigned short lds_b_hi[BN * LDSTR];  // [n][k]
  __shared__ __align__(16) unsigned short lds_b_lo[BN * LDSTR];

  const int tid    = threadIdx.x;
  const int lane   = tid & 31;
  const int wid    = tid >> 5;
  const int wave_m = wid >> 2;
  const int wave_n = wid & 3;
  const int msub   = lane & 15;
  const int ksel   = lane >> 4;

  const int bm = blockIdx.x * BM;
  const int bn = blockIdx.y * BN;

  v8f acc[4][2];
#pragma unroll
  for (int i = 0; i < 4; ++i)
#pragma unroll
    for (int j = 0; j < 2; ++j)
#pragma unroll
      for (int e = 0; e < 8; ++e) acc[i][j][e] = 0.0f;

  for (int kt = 0; kt < K_DIM; kt += BK) {
#pragma unroll
    for (int i = 0; i < 4; ++i) {
      const int id = tid + i * 256;
      const int r  = id >> 3;
      const int c  = (id & 7) << 2;
      const float4 v = *reinterpret_cast<const float4*>(
          x + (size_t)(bm + r) * K_DIM + kt + c);
      unsigned int hp0, lp0, hp1, lp1;
      split2(v.x, v.y, hp0, lp0);
      split2(v.z, v.w, hp1, lp1);
      *reinterpret_cast<uint2*>(&lds_a_hi[r * LDSTR + c]) = make_uint2(hp0, hp1);
      *reinterpret_cast<uint2*>(&lds_a_lo[r * LDSTR + c]) = make_uint2(lp0, lp1);
    }
#pragma unroll
    for (int i = 0; i < 4; ++i) {
      const int id = tid + i * 256;
      const int kr = id >> 5;
      const int nc = (id & 31) << 2;
      const float4 v = *reinterpret_cast<const float4*>(
          w + (size_t)(kt + kr) * H_DIM + bn + nc);
      unsigned short h, l;
      split_bf16_t(v.x, h, l); lds_b_hi[(nc + 0) * LDSTR + kr] = h; lds_b_lo[(nc + 0) * LDSTR + kr] = l;
      split_bf16_t(v.y, h, l); lds_b_hi[(nc + 1) * LDSTR + kr] = h; lds_b_lo[(nc + 1) * LDSTR + kr] = l;
      split_bf16_t(v.z, h, l); lds_b_hi[(nc + 2) * LDSTR + kr] = h; lds_b_lo[(nc + 2) * LDSTR + kr] = l;
      split_bf16_t(v.w, h, l); lds_b_hi[(nc + 3) * LDSTR + kr] = h; lds_b_lo[(nc + 3) * LDSTR + kr] = l;
    }
    __syncthreads();

    FragAB bhi[2], blo[2];
#pragma unroll
    for (int fn = 0; fn < 2; ++fn) {
      const int o = ((wave_n * 2 + fn) * 16 + msub) * LDSTR + ksel * 16;
      bhi[fn].u[0] = *reinterpret_cast<const uint4*>(&lds_b_hi[o]);
      bhi[fn].u[1] = *reinterpret_cast<const uint4*>(&lds_b_hi[o + 8]);
      blo[fn].u[0] = *reinterpret_cast<const uint4*>(&lds_b_lo[o]);
      blo[fn].u[1] = *reinterpret_cast<const uint4*>(&lds_b_lo[o + 8]);
    }
#pragma unroll
    for (int fm = 0; fm < 4; ++fm) {
      const int o0 = ((wave_m * 4 + fm) * 16 + msub) * LDSTR + ksel * 16;
      FragAB ahi, alo;
      ahi.u[0] = *reinterpret_cast<const uint4*>(&lds_a_hi[o0]);
      ahi.u[1] = *reinterpret_cast<const uint4*>(&lds_a_hi[o0 + 8]);
      alo.u[0] = *reinterpret_cast<const uint4*>(&lds_a_lo[o0]);
      alo.u[1] = *reinterpret_cast<const uint4*>(&lds_a_lo[o0 + 8]);
#pragma unroll
      for (int fn = 0; fn < 2; ++fn) {
        acc[fm][fn] = wmma_bf16(ahi.bf, bhi[fn].bf, acc[fm][fn]);
        acc[fm][fn] = wmma_bf16(alo.bf, bhi[fn].bf, acc[fm][fn]);
        acc[fm][fn] = wmma_bf16(ahi.bf, blo[fn].bf, acc[fm][fn]);
      }
    }
    __syncthreads();
  }

#pragma unroll
  for (int fm = 0; fm < 4; ++fm) {
#pragma unroll
    for (int fn = 0; fn < 2; ++fn) {
      const int col = bn + (wave_n * 2 + fn) * 16 + msub;
      const float bv = bias[col];
#pragma unroll
      for (int j = 0; j < 8; ++j) {
        const int row = bm + (wave_m * 4 + fm) * 16 + ksel * 8 + j;
        out[(size_t)row * H_DIM + col] = acc[fm][fn][j] + bv;
      }
    }
  }
}

extern "C" void kernel_launch(void* const* d_in, const int* in_sizes, int n_in,
                              void* d_out, int out_size, void* d_ws, size_t ws_size,
                              hipStream_t stream) {
  const float* x    = (const float*)d_in[0];
  const float* w    = (const float*)d_in[1];
  const float* bias = (const float*)d_in[2];
  float* out        = (float*)d_out;

  const size_t xN = (size_t)M_DIM * K_DIM;   // 33,554,432 elements
  const size_t wN = (size_t)K_DIM * H_DIM;   // 67,108,864 elements
  const size_t need = (xN + wN) * 2 /*hi+lo*/ * 2 /*bytes*/;  // ~403 MB

  dim3 ggrid(M_DIM / BM, H_DIM / BN);  // M fastest -> B panels stay L2-resident
  dim3 gblock(256);

  if (ws_size >= need) {
    unsigned short* xhi = (unsigned short*)d_ws;
    unsigned short* xlo = xhi + xN;
    unsigned short* whi = xlo + xN;
    unsigned short* wlo = whi + wN;
    split_copy_kernel<<<(unsigned)(xN / 1024), 256, 0, stream>>>(x, xhi, xlo);
    split_transpose_kernel<<<dim3(H_DIM / 32, K_DIM / 32), 256, 0, stream>>>(w, whi, wlo);
    gemm_pre_split_async_kernel<<<ggrid, gblock, 0, stream>>>(xhi, xlo, whi, wlo, bias, out);
  } else {
    linear_bias_bf16x3_kernel<<<ggrid, gblock, 0, stream>>>(x, w, bias, out);
  }
}
